// MultiHeadAttention3D_7249904796467
// MI455X (gfx1250) — matrix operand order repaired and checked by reading, not compiled, not run on hardware
//
#include <hip/hip_runtime.h>
#include <hip/hip_bf16.h>
#include <math.h>

// ---------------------------------------------------------------------------
// MultiHeadAttention3D for MI455X (gfx1250, wave32, WMMA + TDM)
// All GEMM-shaped math through v_wmma_f32_16x16x32_bf16; all operand feeds
// (linears AND flash attention K/V chunks) via tensor_load_to_lds with
// double-buffered, bank-conflict-padded LDS tiles.
// ---------------------------------------------------------------------------

typedef __bf16 bf16_t;
typedef __attribute__((ext_vector_type(16))) __bf16 v16bf;
typedef __attribute__((ext_vector_type(8)))  float  v8f;

typedef unsigned int v4u __attribute__((ext_vector_type(4)));
typedef int          v8i __attribute__((ext_vector_type(8)));
typedef int          v4i __attribute__((ext_vector_type(4)));

#define EMBED_DIM 1024
#define NUM_HEADS 16
#define HEAD_DIM  64
#define SEQ       2048
#define BATCH     2
#define MROWS     4096   /* B*N */

// Padded LDS row pitch: 64 data elems + 8 pad elems (=4 dwords pad per 32 dwords)
// -> 144B pitch -> the 16 rows touched by a fragment load map to all 64 banks.
#define LDS_PITCH 72

static constexpr float ATTN_SCALE = 0.125f;  // HEAD_DIM ** -0.5

union Frag16 {
  v16bf v;
  uint4 q[2];
};

// A-matrix (16x32, bf16) fragment per CDNA5 ISA layout:
//   lane l: row M = l%16; element e -> K = k0 + ((l>=16)?8:0) + (e/8)*16 + (e%8)
__device__ __forceinline__ v16bf load_a_frag(const bf16_t* base, int ld,
                                             int row0, int k0, int lane) {
  const bf16_t* p = base + (size_t)(row0 + (lane & 15)) * ld + k0 + ((lane & 16) ? 8 : 0);
  Frag16 f;
  f.q[0] = *(const uint4*)(p);
  f.q[1] = *(const uint4*)(p + 16);
  return f.v;
}

// B-matrix (32x16, bf16) fragment per CDNA5 ISA layout:
//   lane l: col N = col0 + l%16; element e -> K = k0 + ((l>=16)?8:0) + (e/8)*16 + (e%8)
__device__ __forceinline__ v16bf load_b_frag(const bf16_t* base, int ld,
                                             int col0, int k0, int lane) {
  const bf16_t* p = base + (size_t)(col0 + (lane & 15)) * ld + k0 + ((lane & 16) ? 8 : 0);
  Frag16 f;
  f.q[0] = *(const uint4*)(p);
  f.q[1] = *(const uint4*)(p + 16);
  return f.v;
}

__device__ __forceinline__ v8f wmma_bf16f32(v16bf a, v16bf b, v8f c) {
  return __builtin_amdgcn_wmma_f32_16x16x32_bf16(false, a, false, b, (short)0, c,
                                                 false, false);
}

// ---------------------------------------------------------------------------
// Tensor Data Mover: DMA a 2-D bf16 tile (rows x width elems, row stride
// `stride_elems`) from global into LDS at byte offset lds_off, inserting
// 4 dwords of padding after every 32 dwords (i.e. after each 64-elem row),
// matching the LDS_PITCH=72 layout.  D# per CDNA5 ISA 08_async_tensor §8.
// ---------------------------------------------------------------------------
__device__ __forceinline__ void tdm_load_2d(const bf16_t* gptr, unsigned lds_off,
                                            unsigned width_elems, unsigned rows,
                                            unsigned stride_elems) {
  unsigned long long ga = (unsigned long long)(uintptr_t)gptr;

  v4u g0;
  g0[0] = 1u;                                   // count=1, user descriptor
  g0[1] = lds_off;                              // lds_addr (bytes)
  g0[2] = (unsigned)(ga & 0xFFFFFFFFu);         // global_addr[31:0]
  g0[3] = (unsigned)((ga >> 32) & 0x01FFFFFFu)  // global_addr[56:32]
          | (2u << 30);                         // type = 2 ("image")

  const unsigned td0 = 0x40000000u;             // huge tensor dims: tiles are
  const unsigned td1 = 0x40000000u;             // always fully in-bounds
  unsigned long long s0 = (unsigned long long)stride_elems;

  v8i g1;
  g1[0] = (int)((1u << 16)                      // data_size = 2 bytes
                | (1u << 20)                    // pad_enable
                | (4u << 22)                    // pad_interval: 2^(4+1)=32 dwords
                | (3u << 25));                  // pad_amount: 3+1 = 4 dwords
  g1[1] = (int)((td0 & 0xFFFFu) << 16);                       // tensor_dim0[15:0]
  g1[2] = (int)((td0 >> 16) | ((td1 & 0xFFFFu) << 16));       // td0 hi | td1 lo
  g1[3] = (int)((td1 >> 16) | (width_elems << 16));           // td1 hi | tile_dim0
  g1[4] = (int)(rows & 0xFFFFu);                              // tile_dim1, tile_dim2=0
  g1[5] = (int)(s0 & 0xFFFFFFFFu);                            // dim0_stride lo
  g1[6] = (int)((s0 >> 32) & 0xFFFFu);                        // dim0_stride hi
  g1[7] = 0;

  v4i gz = {};
#if __clang_major__ >= 23
  v8i gz8 = {};
  __builtin_amdgcn_tensor_load_to_lds(g0, g1, gz, gz, gz8, 0);
#else
  __builtin_amdgcn_tensor_load_to_lds(g0, g1, gz, gz, 0);
#endif
}

__device__ __forceinline__ unsigned lds_off_of(const void* p) {
  // Generic (flat) pointers to LDS carry the LDS byte offset in the low 32
  // bits (CDNA5 ISA 10.2 aperture mapping).
  return (unsigned)(uintptr_t)p;
}

// ---------------------------------------------------------------------------
// fp32 -> bf16 elementwise conversion
// ---------------------------------------------------------------------------
__global__ __launch_bounds__(256) void cvt_f32_bf16_kernel(
    const float* __restrict__ src, bf16_t* __restrict__ dst, int n) {
  int i = blockIdx.x * 256 + threadIdx.x;
  if (i < n) dst[i] = (bf16_t)src[i];
}

// bf16 transpose: dst[c*rows + r] = src[r*cols + c]
__global__ __launch_bounds__(256) void transpose_bf16_kernel(
    const bf16_t* __restrict__ src, bf16_t* __restrict__ dst, int rows, int cols) {
  int idx = blockIdx.x * 256 + threadIdx.x;
  if (idx < rows * cols) {
    int r = idx / cols, c = idx % cols;
    dst[(size_t)c * rows + r] = src[idx];
  }
}

// ---------------------------------------------------------------------------
// Generic bf16 WMMA GEMM with TDM-fed, double-buffered LDS tiles:
//   Y[m,o] = (sum_k X[m,k]*W[o,k]) * scale + bias[o]
// 256 threads = 8 waves; block tile 128(M) x 64(N); K-step 64.
// ---------------------------------------------------------------------------
__global__ __launch_bounds__(256) void gemm_bf16_kernel(
    const bf16_t* __restrict__ X, int ldx,
    const bf16_t* __restrict__ Wt, int ldw,
    const float* __restrict__ bias,
    float* __restrict__ Yf, int accY,
    bf16_t* __restrict__ Yb,
    int ldy, int K, float scale) {
  const int lane = threadIdx.x & 31;
  const int wave = threadIdx.x >> 5;
  const int wm = wave & 3;   // 4 waves along M
  const int wn = wave >> 2;  // 2 waves along N
  const int mblk = blockIdx.x * 128;
  const int nblk = blockIdx.y * 64;

  __shared__ __align__(16) bf16_t Xs[2][128][LDS_PITCH];  // 36 KiB
  __shared__ __align__(16) bf16_t Ws[2][64][LDS_PITCH];   // 18 KiB

  const bf16_t* Xtile = X + (size_t)mblk * ldx;
  const bf16_t* Wtile = Wt + (size_t)nblk * ldw;

  if (wave == 0) {
    tdm_load_2d(Xtile, lds_off_of(&Xs[0][0][0]), 64, 128, (unsigned)ldx);
    tdm_load_2d(Wtile, lds_off_of(&Ws[0][0][0]), 64, 64, (unsigned)ldw);
  }

  v8f acc00 = {}, acc01 = {}, acc10 = {}, acc11 = {};

  for (int kb = 0; kb < K; kb += 64) {
    const int buf = (kb >> 6) & 1;
    const bool more = (kb + 64) < K;
    if (wave == 0) {
      if (more) {
        tdm_load_2d(Xtile + (kb + 64), lds_off_of(&Xs[buf ^ 1][0][0]), 64, 128,
                    (unsigned)ldx);
        tdm_load_2d(Wtile + (kb + 64), lds_off_of(&Ws[buf ^ 1][0][0]), 64, 64,
                    (unsigned)ldw);
        __builtin_amdgcn_s_wait_tensorcnt(2);  // current buffer's 2 DMAs done
      } else {
        __builtin_amdgcn_s_wait_tensorcnt(0);
      }
    }
    __syncthreads();

#pragma unroll
    for (int kk = 0; kk < 64; kk += 32) {
      v16bf a0 = load_a_frag(&Xs[buf][0][0], LDS_PITCH, wm * 32, kk, lane);
      v16bf a1 = load_a_frag(&Xs[buf][0][0], LDS_PITCH, wm * 32 + 16, kk, lane);
      v16bf b0 = load_b_frag(&Ws[buf][0][0], LDS_PITCH, wn * 32, kk, lane);
      v16bf b1 = load_b_frag(&Ws[buf][0][0], LDS_PITCH, wn * 32 + 16, kk, lane);
      acc00 = wmma_bf16f32(a0, b0, acc00);
      acc01 = wmma_bf16f32(a0, b1, acc01);
      acc10 = wmma_bf16f32(a1, b0, acc10);
      acc11 = wmma_bf16f32(a1, b1, acc11);
    }
    __syncthreads();
  }

  // Epilogue. C/D layout: lane l, VGPR r -> (M = r + (l>=16)*8, N = l%16).
  const int r8 = (lane & 16) ? 8 : 0;
  const int m0 = mblk + wm * 32;
  const int n0 = nblk + wn * 32;
  const int cn0 = n0 + (lane & 15);
  const float bv0 = bias ? bias[cn0] : 0.0f;
  const float bv1 = bias ? bias[cn0 + 16] : 0.0f;

  v8f* accs[2][2] = {{&acc00, &acc01}, {&acc10, &acc11}};
#pragma unroll
  for (int i = 0; i < 2; ++i) {
#pragma unroll
    for (int j = 0; j < 2; ++j) {
      v8f a = *accs[i][j];
#pragma unroll
      for (int r = 0; r < 8; ++r) {
        const int row = m0 + i * 16 + r + r8;
        const int col = n0 + j * 16 + (lane & 15);
        float v = a[r] * scale + (j ? bv1 : bv0);
        if (Yf) {
          size_t off = (size_t)row * ldy + col;
          if (accY) Yf[off] += v; else Yf[off] = v;
        }
        if (Yb) Yb[(size_t)row * ldy + col] = (bf16_t)v;
      }
    }
  }
}

// ---------------------------------------------------------------------------
// Flash-style per-head self-attention with TDM-staged, double-buffered K/V.
// grid = (B*H, N/64), block = 128 (4 waves). Each wave owns 16 query rows.
// Per 64-key chunk: TDM loads K (64x64) and V (64x64) chunks (stride 1024)
// into padded LDS; V is transposed LDS->LDS; scores and P.V run on WMMA with
// online softmax; O stays in WMMA accumulators.
// ---------------------------------------------------------------------------
__global__ __launch_bounds__(128) void self_attn_kernel(
    const bf16_t* __restrict__ Q, const bf16_t* __restrict__ Km,
    const bf16_t* __restrict__ Vm, bf16_t* __restrict__ O) {
  const int lane = threadIdx.x & 31;
  const int wave = threadIdx.x >> 5;
  const int bh = blockIdx.x;
  const int b = bh >> 4, h = bh & 15;
  const int q0 = blockIdx.y * 64;
  const size_t rowbase = (size_t)b * SEQ;
  const int c0 = h * HEAD_DIM;

  __shared__ __align__(16) bf16_t Ks[2][64][LDS_PITCH];  // 18 KiB
  __shared__ __align__(16) bf16_t Vs[2][64][LDS_PITCH];  // 18 KiB
  __shared__ __align__(16) bf16_t Vt[64][LDS_PITCH];     //  9 KiB
  __shared__ __align__(16) float  S[4][16][64];          // 16 KiB
  __shared__ __align__(16) bf16_t P[4][16][64];          //  8 KiB
  __shared__ float rmax[64], rsum[64], ralpha[64];

  if (threadIdx.x < 64) {
    rmax[threadIdx.x] = -3.0e38f;
    rsum[threadIdx.x] = 0.0f;
  }

  const bf16_t* Qbase = Q + rowbase * EMBED_DIM + c0;
  const bf16_t* Kbase = Km + rowbase * EMBED_DIM + c0;
  const bf16_t* Vbase = Vm + rowbase * EMBED_DIM + c0;
  const int mrow = q0 + wave * 16;
  v16bf aq0 = load_a_frag(Qbase, EMBED_DIM, mrow, 0, lane);
  v16bf aq1 = load_a_frag(Qbase, EMBED_DIM, mrow, 32, lane);

  v8f o0 = {}, o1 = {}, o2 = {}, o3 = {};
  v8f* ot[4] = {&o0, &o1, &o2, &o3};
  const int r8 = (lane & 16) ? 8 : 0;

  // Prologue: stage K/V chunk 0.
  if (wave == 0) {
    tdm_load_2d(Kbase, lds_off_of(&Ks[0][0][0]), 64, 64, EMBED_DIM);
    tdm_load_2d(Vbase, lds_off_of(&Vs[0][0][0]), 64, 64, EMBED_DIM);
  }
  __syncthreads();

  for (int nb = 0; nb < SEQ; nb += 64) {
    const int buf = (nb >> 6) & 1;
    const bool more = (nb + 64) < SEQ;
    if (wave == 0) {
      if (more) {
        tdm_load_2d(Kbase + (size_t)(nb + 64) * EMBED_DIM,
                    lds_off_of(&Ks[buf ^ 1][0][0]), 64, 64, EMBED_DIM);
        tdm_load_2d(Vbase + (size_t)(nb + 64) * EMBED_DIM,
                    lds_off_of(&Vs[buf ^ 1][0][0]), 64, 64, EMBED_DIM);
        __builtin_amdgcn_s_wait_tensorcnt(2);
      } else {
        __builtin_amdgcn_s_wait_tensorcnt(0);
      }
    }
    __syncthreads();  // staged chunk visible to all 4 waves

    // Transpose V chunk LDS->LDS: read 8 contiguous d-values (b128), write
    // 8 strided b16. 512 tasks across 128 threads.
    for (int idx = threadIdx.x; idx < 64 * 8; idx += 128) {
      const int n = idx >> 3, dc = (idx & 7) * 8;
      uint4 vv = *(const uint4*)&Vs[buf][n][dc];
      const bf16_t* e = (const bf16_t*)&vv;
#pragma unroll
      for (int j = 0; j < 8; ++j) Vt[dc + j][n] = e[j];
    }

    // Scores: S = Q (16x64) x K^T from staged Ks.
#pragma unroll
    for (int j = 0; j < 4; ++j) {
      v16bf b0 = load_b_frag(&Ks[buf][0][0], LDS_PITCH, j * 16, 0, lane);
      v16bf b1 = load_b_frag(&Ks[buf][0][0], LDS_PITCH, j * 16, 32, lane);
      v8f s = {};
      s = wmma_bf16f32(aq0, b0, s);
      s = wmma_bf16f32(aq1, b1, s);
#pragma unroll
      for (int r = 0; r < 8; ++r)
        S[wave][r + r8][j * 16 + (lane & 15)] = s[r] * ATTN_SCALE;
    }
    __syncthreads();  // S complete; Vt complete

    // Online softmax: lane<16 owns one of this wave's 16 rows.
    if (lane < 16) {
      const int row = wave * 16 + lane;
      float mx = -3.0e38f;
#pragma unroll 8
      for (int c = 0; c < 64; ++c) mx = fmaxf(mx, S[wave][lane][c]);
      const float mo = rmax[row];
      const float mn = fmaxf(mo, mx);
      const float al = __expf(mo - mn);
      float sm = 0.0f;
#pragma unroll 8
      for (int c = 0; c < 64; ++c) {
        float p = __expf(S[wave][lane][c] - mn);
        P[wave][lane][c] = (bf16_t)p;
        sm += p;
      }
      rsum[row] = rsum[row] * al + sm;
      rmax[row] = mn;
      ralpha[row] = al;
    }
    __syncthreads();

    // Rescale running O accumulators.
#pragma unroll
    for (int t = 0; t < 4; ++t) {
      v8f v = *ot[t];
#pragma unroll
      for (int r = 0; r < 8; ++r) v[r] *= ralpha[wave * 16 + r + r8];
      *ot[t] = v;
    }

    // O += P (16x64) x V (64x64) from transposed Vt.
    v16bf ap0 = load_a_frag(&P[wave][0][0], 64, 0, 0, lane);
    v16bf ap1 = load_a_frag(&P[wave][0][0], 64, 0, 32, lane);
#pragma unroll
    for (int t = 0; t < 4; ++t) {
      v16bf bv0 = load_b_frag(&Vt[0][0], LDS_PITCH, t * 16, 0, lane);
      v16bf bv1 = load_b_frag(&Vt[0][0], LDS_PITCH, t * 16, 32, lane);
      v8f v = *ot[t];
      v = wmma_bf16f32(ap0, bv0, v);
      v = wmma_bf16f32(ap1, bv1, v);
      *ot[t] = v;
    }
    __syncthreads();  // all reads of Ks/Vs/Vt done before next stage/rebuild
  }

#pragma unroll
  for (int t = 0; t < 4; ++t) {
    v8f v = *ot[t];
#pragma unroll
    for (int r = 0; r < 8; ++r) {
      const int row = wave * 16 + r + r8;
      const float inv = 1.0f / rsum[row];
      O[(rowbase + q0 + row) * EMBED_DIM + c0 + t * 16 + (lane & 15)] =
          (bf16_t)(v[r] * inv);
    }
  }
}

// ---------------------------------------------------------------------------
// Row softmax (fp32 in, bf16 probabilities out). One block per row.
// ---------------------------------------------------------------------------
__global__ __launch_bounds__(256) void softmax_bf16_kernel(
    const float* __restrict__ S, bf16_t* __restrict__ P, int cols) {
  __shared__ float red[256];
  const int row = blockIdx.x;
  const float* s = S + (size_t)row * cols;
  bf16_t* p = P + (size_t)row * cols;

  float mx = -3.0e38f;
  for (int c = threadIdx.x; c < cols; c += 256) mx = fmaxf(mx, s[c]);
  red[threadIdx.x] = mx;
  __syncthreads();
  for (int off = 128; off > 0; off >>= 1) {
    if (threadIdx.x < off) red[threadIdx.x] = fmaxf(red[threadIdx.x], red[threadIdx.x + off]);
    __syncthreads();
  }
  mx = red[0];
  __syncthreads();

  float sm = 0.0f;
  for (int c = threadIdx.x; c < cols; c += 256) sm += __expf(s[c] - mx);
  red[threadIdx.x] = sm;
  __syncthreads();
  for (int off = 128; off > 0; off >>= 1) {
    if (threadIdx.x < off) red[threadIdx.x] += red[threadIdx.x + off];
    __syncthreads();
  }
  const float inv = 1.0f / red[0];

  for (int c = threadIdx.x; c < cols; c += 256)
    p[c] = (bf16_t)(__expf(s[c] - mx) * inv);
}

// ---------------------------------------------------------------------------
// Host launch
// ---------------------------------------------------------------------------
extern "C" void kernel_launch(void* const* d_in, const int* in_sizes, int n_in,
                              void* d_out, int out_size, void* d_ws, size_t ws_size,
                              hipStream_t stream) {
  (void)in_sizes; (void)n_in; (void)out_size; (void)ws_size;

  const float* x   = (const float*)d_in[0];
  const float* ctx = (const float*)d_in[1];
  const float* Wsrc[7];  // q,k,v,o,cq,ck,cv
  const float* bsrc[7];
  for (int i = 0; i < 7; ++i) {
    Wsrc[i] = (const float*)d_in[2 + 2 * i];
    bsrc[i] = (const float*)d_in[3 + 2 * i];
  }
  float* out = (float*)d_out;

  const size_t MB = 1ull << 20;
  char* ws = (char*)d_ws;

  bf16_t* xb    = (bf16_t*)(ws + 0 * MB);    // 8 MiB (dead after q/k/v GEMMs)
  bf16_t* ctxb  = (bf16_t*)(ws + 8 * MB);    // 8 MiB (dead after ck/cv GEMMs)
  bf16_t* Wb[7];
  for (int i = 0; i < 7; ++i) Wb[i] = (bf16_t*)(ws + (16 + 2 * i) * MB);
  bf16_t* qb    = (bf16_t*)(ws + 30 * MB);
  bf16_t* kb    = (bf16_t*)(ws + 38 * MB);
  bf16_t* vb    = (bf16_t*)(ws + 46 * MB);
  bf16_t* attnb = (bf16_t*)(ws + 54 * MB);
  bf16_t* aob   = (bf16_t*)(ws + 62 * MB);
  bf16_t* cqb   = (bf16_t*)(ws + 70 * MB);
  bf16_t* ckb   = (bf16_t*)(ws + 78 * MB);
  bf16_t* cvb   = (bf16_t*)(ws + 86 * MB);
  bf16_t* cvT   = (bf16_t*)(ws + 94 * MB);
  float*  Sx    = (float*)(ws + 30 * MB);    // 32 MiB, aliases qb..attnb (dead)
  bf16_t* Pb    = (bf16_t*)(ws + 0 * MB);    // 16 MiB, aliases xb+ctxb (dead)

  const int nxc = MROWS * EMBED_DIM;
  const int nw  = EMBED_DIM * EMBED_DIM;

  cvt_f32_bf16_kernel<<<(nxc + 255) / 256, 256, 0, stream>>>(x, xb, nxc);
  cvt_f32_bf16_kernel<<<(nxc + 255) / 256, 256, 0, stream>>>(ctx, ctxb, nxc);
  for (int i = 0; i < 7; ++i)
    cvt_f32_bf16_kernel<<<(nw + 255) / 256, 256, 0, stream>>>(Wsrc[i], Wb[i], nw);

  const dim3 gLin(MROWS / 128, EMBED_DIM / 64);  // (32,16)

  gemm_bf16_kernel<<<gLin, 256, 0, stream>>>(xb, EMBED_DIM, Wb[0], EMBED_DIM, bsrc[0],
                                             nullptr, 0, qb, EMBED_DIM, EMBED_DIM, 1.0f);
  gemm_bf16_kernel<<<gLin, 256, 0, stream>>>(xb, EMBED_DIM, Wb[1], EMBED_DIM, bsrc[1],
                                             nullptr, 0, kb, EMBED_DIM, EMBED_DIM, 1.0f);
  gemm_bf16_kernel<<<gLin, 256, 0, stream>>>(xb, EMBED_DIM, Wb[2], EMBED_DIM, bsrc[2],
                                             nullptr, 0, vb, EMBED_DIM, EMBED_DIM, 1.0f);

  self_attn_kernel<<<dim3(BATCH * NUM_HEADS, SEQ / 64), 128, 0, stream>>>(qb, kb, vb, attnb);

  gemm_bf16_kernel<<<gLin, 256, 0, stream>>>(attnb, EMBED_DIM, Wb[3], EMBED_DIM, bsrc[3],
                                             out, 0, aob, EMBED_DIM, EMBED_DIM, 1.0f);

  gemm_bf16_kernel<<<gLin, 256, 0, stream>>>(aob, EMBED_DIM, Wb[4], EMBED_DIM, bsrc[4],
                                             nullptr, 0, cqb, EMBED_DIM, EMBED_DIM, 1.0f);
  gemm_bf16_kernel<<<gLin, 256, 0, stream>>>(ctxb, EMBED_DIM, Wb[5], EMBED_DIM, bsrc[5],
                                             nullptr, 0, ckb, EMBED_DIM, EMBED_DIM, 1.0f);
  gemm_bf16_kernel<<<gLin, 256, 0, stream>>>(ctxb, EMBED_DIM, Wb[6], EMBED_DIM, bsrc[6],
                                             nullptr, 0, cvb, EMBED_DIM, EMBED_DIM, 1.0f);

  for (int b = 0; b < BATCH; ++b)
    transpose_bf16_kernel<<<(SEQ * EMBED_DIM + 255) / 256, 256, 0, stream>>>(
        cvb + (size_t)b * SEQ * EMBED_DIM, cvT + (size_t)b * EMBED_DIM * SEQ,
        SEQ, EMBED_DIM);

  for (int b = 0; b < BATCH; ++b)
    gemm_bf16_kernel<<<dim3(SEQ / 128, SEQ / 64), 256, 0, stream>>>(
        cqb + (size_t)b * SEQ * EMBED_DIM, EMBED_DIM,
        ckb + (size_t)b * SEQ * EMBED_DIM, EMBED_DIM, nullptr,
        Sx + (size_t)b * SEQ * SEQ, 0, nullptr, SEQ, EMBED_DIM, ATTN_SCALE);

  softmax_bf16_kernel<<<BATCH * SEQ, 256, 0, stream>>>(Sx, Pb, SEQ);

  for (int b = 0; b < BATCH; ++b)
    gemm_bf16_kernel<<<dim3(SEQ / 128, EMBED_DIM / 64), 256, 0, stream>>>(
        Pb + (size_t)b * SEQ * SEQ, SEQ,
        cvT + (size_t)b * EMBED_DIM * SEQ, SEQ, nullptr,
        out + (size_t)b * SEQ * EMBED_DIM, 1, nullptr, EMBED_DIM, SEQ, 1.0f);
}
